// ProteinTransformer_48404281426217
// MI455X (gfx1250) — compile-verified, matrix operand order checked by reading, not run on hardware
//
#include <hip/hip_runtime.h>
#include <cstddef>
#include <cstdint>

// ---------------------------------------------------------------- constants
#define BD   4          // batch
#define LSEQ 2048       // seq len
#define DMOD 1024       // d_model
#define HN   16         // heads
#define DH   64         // head dim
#define DFF  4096
#define NLAY 4
#define BS   128        // attention block size
#define NB   16         // blocks per sequence
#define MTOT (BD * LSEQ)  // 8192 rows

typedef __attribute__((ext_vector_type(16))) __bf16       v16bf;
typedef __attribute__((ext_vector_type(8)))  float        v8f;
typedef __attribute__((ext_vector_type(4)))  unsigned int u32x4;

#define ZERO8 {0.f,0.f,0.f,0.f,0.f,0.f,0.f,0.f}

// ---------------------------------------------------------------- helpers
__device__ __forceinline__ v16bf load_frag(const __bf16* p) {
    // 16-bit operand fragment: two contiguous 16B runs per lane
    union { v16bf v; u32x4 u[2]; } f;
    f.u[0] = *(const u32x4*)(p);
    f.u[1] = *(const u32x4*)(p + 16);
    return f.v;
}

__device__ __forceinline__ v8f wmma_bf16(v16bf a, v16bf b, v8f c) {
    return __builtin_amdgcn_wmma_f32_16x16x32_bf16(
        /*neg_a=*/false, a, /*neg_b=*/false, b,
        /*c_mod=*/(short)0, c, /*reuse_a=*/false, /*reuse_b=*/false);
}

// CDNA5 async global->LDS copy (16B per lane, ASYNCcnt-tracked, no VGPR round trip)
__device__ __forceinline__ void async_cp16_raw(unsigned lds_off, unsigned long long ga) {
    asm volatile("global_load_async_to_lds_b128 %0, %1, off"
                 :: "v"(lds_off), "v"(ga) : "memory");
}
__device__ __forceinline__ void async_cp16(const __bf16* lds_dst, const __bf16* gsrc) {
    async_cp16_raw((unsigned)(uintptr_t)(const void*)lds_dst,
                   (unsigned long long)(uintptr_t)gsrc);
}
#define WAIT_ASYNC(n) asm volatile("s_wait_asynccnt " #n ::: "memory")

__device__ __forceinline__ float block_sum256(float v) {
    __shared__ float sb[9];
#pragma unroll
    for (int m = 16; m; m >>= 1) v += __shfl_xor(v, m, 32);
    if ((threadIdx.x & 31) == 0) sb[threadIdx.x >> 5] = v;
    __syncthreads();
    if (threadIdx.x == 0) {
        float t = 0.f;
#pragma unroll
        for (int i = 0; i < 8; ++i) t += sb[i];
        sb[8] = t;
    }
    __syncthreads();
    float r = sb[8];
    __syncthreads();
    return r;
}

// ---------------------------------------------------------------- fp32 [K][N] -> bf16 transposed [N][K]
// LDS-tiled 64x64 transpose; blockIdx.z selects the layer slice.
__global__ __launch_bounds__(256) void cvt_t_kernel(const float* __restrict__ in,
                                                    __bf16* __restrict__ out,
                                                    int K, int N) {
    __shared__ __bf16 tl[64][65];
    in  += (size_t)blockIdx.z * K * N;
    out += (size_t)blockIdx.z * K * N;
    const int kb = blockIdx.y * 64;
    const int nb = blockIdx.x * 64;
#pragma unroll
    for (int j = 0; j < 16; ++j) {
        int idx = threadIdx.x + j * 256;        // 0..4095
        int r = idx >> 6, c = idx & 63;
        tl[r][c] = (__bf16)in[(size_t)(kb + r) * N + nb + c];
    }
    __syncthreads();
#pragma unroll
    for (int j = 0; j < 16; ++j) {
        int idx = threadIdx.x + j * 256;
        int r = idx >> 6, c = idx & 63;
        out[(size_t)(nb + r) * K + kb + c] = tl[c][r];
    }
}

// ---------------------------------------------------------------- embedding gather
__global__ __launch_bounds__(256) void embed_kernel(const int* __restrict__ ids,
                                                    const float* __restrict__ tok,
                                                    const float* __restrict__ pos,
                                                    float* __restrict__ y) {
    int row = blockIdx.x;                 // 0..MTOT-1  (b*L + l)
    int l   = row & (LSEQ - 1);
    int id  = ids[row];
    size_t base = (size_t)row * DMOD;
#pragma unroll
    for (int j = 0; j < 4; ++j) {
        int idx = threadIdx.x + j * 256;
        y[base + idx] = tok[(size_t)id * DMOD + idx] + pos[(size_t)l * DMOD + idx];
    }
}

// ---------------------------------------------------------------- LayerNorm (+optional residual)
__global__ __launch_bounds__(256) void ln_kernel(const float* __restrict__ src,
                                                 const float* __restrict__ resid,
                                                 const float* __restrict__ g,
                                                 const float* __restrict__ bta,
                                                 float* __restrict__ xo,
                                                 __bf16* __restrict__ xb) {
    int row = blockIdx.x;
    size_t base = (size_t)row * DMOD;
    float v[4];
    float s = 0.f;
#pragma unroll
    for (int j = 0; j < 4; ++j) {
        int idx = threadIdx.x + j * 256;
        float t = src[base + idx];
        if (resid) t += resid[base + idx];
        v[j] = t; s += t;
    }
    float mean = block_sum256(s) * (1.0f / DMOD);
    float ss = 0.f;
#pragma unroll
    for (int j = 0; j < 4; ++j) { v[j] -= mean; ss += v[j] * v[j]; }
    float var = block_sum256(ss) * (1.0f / DMOD);
    float r = rsqrtf(var + 1e-5f);
#pragma unroll
    for (int j = 0; j < 4; ++j) {
        int idx = threadIdx.x + j * 256;
        float o = v[j] * r * g[idx] + bta[idx];
        xo[base + idx] = o;
        xb[base + idx] = (__bf16)o;
    }
}

// ---------------------------------------------------------------- WMMA GEMM  C = A(bf16) * Bt(bf16)^T [+bias][relu]
// Bt is pre-transposed [N][K] row-major. 128x128 block tile, K-step 32,
// double-buffered async global->LDS staging with peeled pipeline epilogue.
// vstore!=0: write Cb into per-(block,head)-transposed V layout [BH][64][128].
__global__ __launch_bounds__(256) void gemm_bf16_kernel(
        const __bf16* __restrict__ A, const __bf16* __restrict__ Bt,
        const float* __restrict__ bias, float* __restrict__ Cf, __bf16* __restrict__ Cb,
        int Mdim, int Ndim, int Kdim, int relu, int vstore) {
    constexpr int LDT = 40;                               // padded pitch (bf16 elems)
    constexpr unsigned BUFB = 128 * LDT * 2;              // bytes per LDS buffer (10240)
    __shared__ __align__(16) __bf16 Alds[2][128 * LDT];
    __shared__ __align__(16) __bf16 Blds[2][128 * LDT];   // Bt tile: [n][k]

    const int tid  = threadIdx.x;
    const int lane = tid & 31;
    const int hlf  = lane >> 4;
    const int r16  = lane & 15;
    const int wv   = tid >> 5;
    const int waveM = (wv >> 1) * 32;
    const int waveN = (wv & 1) * 64;
    const int blockM = blockIdx.y * 128;
    const int blockN = blockIdx.x * 128;

    v8f acc[2][4];
#pragma unroll
    for (int mt = 0; mt < 2; ++mt)
#pragma unroll
        for (int nt = 0; nt < 4; ++nt) { v8f z = ZERO8; acc[mt][nt] = z; }

    // precompute per-thread staging addresses (2 chunks each for A and Bt)
    const int r0  = (tid * 2) >> 2,     kc0 = (tid * 2) & 3;
    const int r1  = (tid * 2 + 1) >> 2, kc1 = (tid * 2 + 1) & 3;
    const unsigned long long aS0 =
        (unsigned long long)(uintptr_t)(A  + (size_t)(blockM + r0) * Kdim + kc0 * 8);
    const unsigned long long aS1 =
        (unsigned long long)(uintptr_t)(A  + (size_t)(blockM + r1) * Kdim + kc1 * 8);
    const unsigned long long bS0 =
        (unsigned long long)(uintptr_t)(Bt + (size_t)(blockN + r0) * Kdim + kc0 * 8);
    const unsigned long long bS1 =
        (unsigned long long)(uintptr_t)(Bt + (size_t)(blockN + r1) * Kdim + kc1 * 8);
    const unsigned aD0 = (unsigned)(uintptr_t)&Alds[0][r0 * LDT + kc0 * 8];
    const unsigned aD1 = (unsigned)(uintptr_t)&Alds[0][r1 * LDT + kc1 * 8];
    const unsigned bD0 = (unsigned)(uintptr_t)&Blds[0][r0 * LDT + kc0 * 8];
    const unsigned bD1 = (unsigned)(uintptr_t)&Blds[0][r1 * LDT + kc1 * 8];

    auto stage = [&](unsigned bufoff, unsigned long long koff) {
        async_cp16_raw(aD0 + bufoff, aS0 + koff);
        async_cp16_raw(aD1 + bufoff, aS1 + koff);
        async_cp16_raw(bD0 + bufoff, bS0 + koff);
        async_cp16_raw(bD1 + bufoff, bS1 + koff);
    };
    auto compute = [&](int cur) {
        const __bf16* Ab = Alds[cur];
        const __bf16* Bb = Blds[cur];
        v16bf af0 = load_frag(&Ab[(waveM + 0 * 16 + r16) * LDT + hlf * 8]);
        v16bf af1 = load_frag(&Ab[(waveM + 1 * 16 + r16) * LDT + hlf * 8]);
#pragma unroll
        for (int nt = 0; nt < 4; ++nt) {
            v16bf bf = load_frag(&Bb[(waveN + nt * 16 + r16) * LDT + hlf * 8]);
            acc[0][nt] = wmma_bf16(af0, bf, acc[0][nt]);
            acc[1][nt] = wmma_bf16(af1, bf, acc[1][nt]);
        }
    };

    const int T = Kdim >> 5;                   // K-steps of 32 (64 bytes)
    stage(0, 0);
    int t = 0;
    for (; t + 1 < T; ++t) {                   // steady state: unconditional wait 4
        stage(((t + 1) & 1) ? BUFB : 0u, (unsigned long long)(t + 1) * 64);
        WAIT_ASYNC(0x4);                       // tile t's 4 copies complete (in-order)
        __syncthreads();
        compute(t & 1);
        __syncthreads();                       // all waves done with buf before reuse
    }
    WAIT_ASYNC(0x0);                           // epilogue: last tile
    __syncthreads();
    compute(t & 1);

#pragma unroll
    for (int mt = 0; mt < 2; ++mt)
#pragma unroll
        for (int nt = 0; nt < 4; ++nt) {
            int col = blockN + waveN + nt * 16 + r16;
            float bvv = bias ? bias[col] : 0.f;
#pragma unroll
            for (int i = 0; i < 8; ++i) {
                int row = blockM + waveM + mt * 16 + hlf * 8 + i;   // C layout: vgpr i -> row
                float v = acc[mt][nt][i] + bvv;
                if (relu) v = fmaxf(v, 0.f);
                if (Cf) Cf[(size_t)row * Ndim + col] = v;
                if (Cb) {
                    size_t off;
                    if (vstore) {
                        int bh = (row >> 7) * HN + (col >> 6);      // (b*NB+nb)*HN + h
                        off = ((size_t)bh * DH + (col & 63)) * BS + (row & 127);
                    } else {
                        off = (size_t)row * Ndim + col;
                    }
                    Cb[off] = (__bf16)v;
                }
            }
        }
}

// ---------------------------------------------------------------- block attention (one WG per b,block,head)
// vt is pre-transposed: [BH][DH=64][BS=128]
__global__ __launch_bounds__(256) void attn_kernel(const __bf16* __restrict__ qg,
                                                   const __bf16* __restrict__ kg,
                                                   const __bf16* __restrict__ vt,
                                                   const int* __restrict__ ids,
                                                   __bf16* __restrict__ ctx) {
    constexpr int LDQ = 72;     // pitch for q/k tiles [128][72]
    constexpr int LDP = 136;    // pitch for probs [128][136] and vT [64][136]
    __shared__ __align__(16) char smem[2 * 128 * LDQ * 2 + 64 * LDP * 2 + 128 * 4];
    __bf16* Q  = (__bf16*)smem;                 // [128][72]
    __bf16* Kl = Q  + 128 * LDQ;                // [128][72]
    __bf16* VT = Kl + 128 * LDQ;                // [64][136]
    float*  mk = (float*)(VT + 64 * LDP);       // [128] key mask
    __bf16* P  = Q;                             // probs overlay q+k region (34816 <= 36864 B)

    const int blk = blockIdx.x;                 // b*NB*HN + nb*HN + h
    const int h  = blk % HN;
    const int nb = (blk / HN) % NB;
    const int b  = blk / (HN * NB);
    const int rowbase = b * LSEQ + nb * BS;
    const size_t goff = (size_t)rowbase * DMOD + h * DH;
    const __bf16* vtp = vt + (size_t)((b * NB + nb) * HN + h) * (DH * BS);

    const int tid  = threadIdx.x;
    const int lane = tid & 31;
    const int hlf  = lane >> 4;
    const int r16  = lane & 15;
    const int wv   = tid >> 5;                  // wave owns query rows [wv*16, wv*16+16)

    // async-stage q, k, vT: 12 aligned 16B copies per thread
#pragma unroll
    for (int c = 0; c < 4; ++c) {
        int chunk = tid * 4 + c;                // 0..1023
        int r = chunk >> 3, c8 = chunk & 7;     // 128 rows x 8 chunks-of-8
        async_cp16(&Q[r * LDQ + c8 * 8],  qg + goff + (size_t)r * DMOD + c8 * 8);
        async_cp16(&Kl[r * LDQ + c8 * 8], kg + goff + (size_t)r * DMOD + c8 * 8);
        int d = chunk >> 4, rc = chunk & 15;    // 64 rows x 16 chunks-of-8
        async_cp16(&VT[d * LDP + rc * 8], vtp + (size_t)d * BS + rc * 8);
    }
    if (tid < 128) mk[tid] = (ids[rowbase + tid] != 0) ? 1.f : 0.f;
    WAIT_ASYNC(0x0);
    __syncthreads();

    // scores S = q k^T : per wave 16 rows x 128 cols, K=64 -> 2 WMMA k-steps
    v8f acc[8];
#pragma unroll
    for (int nt = 0; nt < 8; ++nt) { v8f z = ZERO8; acc[nt] = z; }
#pragma unroll
    for (int ks = 0; ks < 2; ++ks) {
        v16bf a = load_frag(&Q[(wv * 16 + r16) * LDQ + ks * 32 + hlf * 8]);
#pragma unroll
        for (int nt = 0; nt < 8; ++nt) {
            v16bf bf = load_frag(&Kl[(nt * 16 + r16) * LDQ + ks * 32 + hlf * 8]);
            acc[nt] = wmma_bf16(a, bf, acc[nt]);
        }
    }

    // scale + key-padding mask (replace with -1e9, like the reference)
    const float scale = 0.125f;                 // 1/sqrt(64)
#pragma unroll
    for (int nt = 0; nt < 8; ++nt) {
        float m = mk[nt * 16 + r16];
#pragma unroll
        for (int i = 0; i < 8; ++i) {
            float s = acc[nt][i] * scale;
            acc[nt][i] = (m != 0.f) ? s : -1e9f;
        }
    }
    // softmax per row; row (i + hlf*8) elements live in the 16 lanes of this half-wave
    float rinv[8];
#pragma unroll
    for (int i = 0; i < 8; ++i) {
        float mx = -3.0e38f;
#pragma unroll
        for (int nt = 0; nt < 8; ++nt) mx = fmaxf(mx, acc[nt][i]);
#pragma unroll
        for (int m = 8; m; m >>= 1) mx = fmaxf(mx, __shfl_xor(mx, m, 32));
        float s = 0.f;
#pragma unroll
        for (int nt = 0; nt < 8; ++nt) {
            float e = __expf(acc[nt][i] - mx);
            acc[nt][i] = e; s += e;
        }
#pragma unroll
        for (int m = 8; m; m >>= 1) s += __shfl_xor(s, m, 32);
        rinv[i] = 1.f / s;
    }
    __syncthreads();                            // all waves done reading Q/K LDS
    // write normalized probs (bf16) into overlay
#pragma unroll
    for (int nt = 0; nt < 8; ++nt)
#pragma unroll
        for (int i = 0; i < 8; ++i) {
            int m = wv * 16 + hlf * 8 + i;
            int n = nt * 16 + r16;
            P[m * LDP + n] = (__bf16)(acc[nt][i] * rinv[i]);
        }
    __syncthreads();

    // ctx = P @ v : per wave 16 rows x 64 cols, K=128 -> 4 WMMA k-steps
    v8f c2[4];
#pragma unroll
    for (int nt = 0; nt < 4; ++nt) { v8f z = ZERO8; c2[nt] = z; }
#pragma unroll
    for (int ks = 0; ks < 4; ++ks) {
        v16bf a = load_frag(&P[(wv * 16 + r16) * LDP + ks * 32 + hlf * 8]);
#pragma unroll
        for (int nt = 0; nt < 4; ++nt) {
            v16bf bf = load_frag(&VT[(nt * 16 + r16) * LDP + ks * 32 + hlf * 8]);
            c2[nt] = wmma_bf16(a, bf, c2[nt]);
        }
    }
#pragma unroll
    for (int nt = 0; nt < 4; ++nt)
#pragma unroll
        for (int i = 0; i < 8; ++i) {
            int m = wv * 16 + hlf * 8 + i;
            int n = nt * 16 + r16;
            ctx[(size_t)(rowbase + m) * DMOD + h * DH + n] = (__bf16)c2[nt][i];
        }
}

// ---------------------------------------------------------------- head: (D)->(3)
__global__ __launch_bounds__(256) void head_kernel(const float* __restrict__ x,
                                                   const float* __restrict__ hW,
                                                   const float* __restrict__ hb,
                                                   float* __restrict__ out) {
    int row = blockIdx.x;
    float p0 = 0.f, p1 = 0.f, p2 = 0.f;
#pragma unroll
    for (int j = 0; j < 4; ++j) {
        int idx = threadIdx.x + j * 256;
        float xv = x[(size_t)row * DMOD + idx];
        p0 += xv * hW[idx * 3 + 0];
        p1 += xv * hW[idx * 3 + 1];
        p2 += xv * hW[idx * 3 + 2];
    }
    p0 = block_sum256(p0);
    p1 = block_sum256(p1);
    p2 = block_sum256(p2);
    if (threadIdx.x == 0) {
        out[row * 3 + 0] = p0 + hb[0];
        out[row * 3 + 1] = p1 + hb[1];
        out[row * 3 + 2] = p2 + hb[2];
    }
}

// ---------------------------------------------------------------- launch
extern "C" void kernel_launch(void* const* d_in, const int* in_sizes, int n_in,
                              void* d_out, int out_size, void* d_ws, size_t ws_size,
                              hipStream_t stream) {
    const int*   ids  = (const int*)  d_in[0];
    const float* tok  = (const float*)d_in[1];
    const float* pos  = (const float*)d_in[2];
    const float* gE   = (const float*)d_in[3];
    const float* bE   = (const float*)d_in[4];
    const float* Wq   = (const float*)d_in[5];
    const float* Wk   = (const float*)d_in[6];
    const float* Wv   = (const float*)d_in[7];
    const float* Wo   = (const float*)d_in[8];
    const float* g1   = (const float*)d_in[9];
    const float* b1   = (const float*)d_in[10];
    const float* W1   = (const float*)d_in[11];
    const float* bf1  = (const float*)d_in[12];
    const float* W2   = (const float*)d_in[13];
    const float* bf2  = (const float*)d_in[14];
    const float* g2   = (const float*)d_in[15];
    const float* b2   = (const float*)d_in[16];
    const float* hW   = (const float*)d_in[17];
    const float* hbv  = (const float*)d_in[18];
    float* out = (float*)d_out;

    char* p = (char*)d_ws;
    auto carve = [&](size_t bytes) -> char* {
        char* r = p;
        p += (bytes + 255) & ~(size_t)255;
        return r;
    };
    float*  x    = (float*) carve((size_t)MTOT * DMOD * 4);
    float*  y    = (float*) carve((size_t)MTOT * DMOD * 4);
    __bf16* xb   = (__bf16*)carve((size_t)MTOT * DMOD * 2);
    __bf16* qb   = (__bf16*)carve((size_t)MTOT * DMOD * 2);
    __bf16* kb   = (__bf16*)carve((size_t)MTOT * DMOD * 2);
    __bf16* vb   = (__bf16*)carve((size_t)MTOT * DMOD * 2);   // [BH][64][128] layout
    __bf16* ctxb = (__bf16*)carve((size_t)MTOT * DMOD * 2);
    __bf16* hbuf = (__bf16*)carve((size_t)MTOT * DFF  * 2);
    __bf16* wqb  = (__bf16*)carve((size_t)NLAY * DMOD * DMOD * 2);   // transposed [N][K]
    __bf16* wkb  = (__bf16*)carve((size_t)NLAY * DMOD * DMOD * 2);
    __bf16* wvb  = (__bf16*)carve((size_t)NLAY * DMOD * DMOD * 2);
    __bf16* wob  = (__bf16*)carve((size_t)NLAY * DMOD * DMOD * 2);
    __bf16* w1b  = (__bf16*)carve((size_t)NLAY * DMOD * DFF  * 2);   // [DFF][D]
    __bf16* w2b  = (__bf16*)carve((size_t)NLAY * DFF  * DMOD * 2);   // [D][DFF]

    // one-time fp32 -> bf16 conversion with transpose (weights become [N][K])
    dim3 gtDD(DMOD / 64, DMOD / 64, NLAY);
    cvt_t_kernel<<<gtDD, 256, 0, stream>>>(Wq, wqb, DMOD, DMOD);
    cvt_t_kernel<<<gtDD, 256, 0, stream>>>(Wk, wkb, DMOD, DMOD);
    cvt_t_kernel<<<gtDD, 256, 0, stream>>>(Wv, wvb, DMOD, DMOD);
    cvt_t_kernel<<<gtDD, 256, 0, stream>>>(Wo, wob, DMOD, DMOD);
    dim3 gtW1(DFF / 64, DMOD / 64, NLAY);
    cvt_t_kernel<<<gtW1, 256, 0, stream>>>(W1, w1b, DMOD, DFF);
    dim3 gtW2(DMOD / 64, DFF / 64, NLAY);
    cvt_t_kernel<<<gtW2, 256, 0, stream>>>(W2, w2b, DFF, DMOD);

    embed_kernel<<<MTOT, 256, 0, stream>>>(ids, tok, pos, y);
    ln_kernel<<<MTOT, 256, 0, stream>>>(y, nullptr, gE, bE, x, xb);

    dim3 gDD(DMOD / 128, MTOT / 128);   // N=1024 tiles
    dim3 gFF(DFF  / 128, MTOT / 128);   // N=4096 tiles
    for (int l = 0; l < NLAY; ++l) {
        const size_t oDD = (size_t)l * DMOD * DMOD;
        const size_t oDF = (size_t)l * DMOD * DFF;
        gemm_bf16_kernel<<<gDD, 256, 0, stream>>>(xb, wqb + oDD, nullptr, nullptr, qb,
                                                  MTOT, DMOD, DMOD, 0, 0);
        gemm_bf16_kernel<<<gDD, 256, 0, stream>>>(xb, wkb + oDD, nullptr, nullptr, kb,
                                                  MTOT, DMOD, DMOD, 0, 0);
        gemm_bf16_kernel<<<gDD, 256, 0, stream>>>(xb, wvb + oDD, nullptr, nullptr, vb,
                                                  MTOT, DMOD, DMOD, 0, 1);
        attn_kernel<<<BD * NB * HN, 256, 0, stream>>>(qb, kb, vb, ids, ctxb);
        gemm_bf16_kernel<<<gDD, 256, 0, stream>>>(ctxb, wob + oDD, nullptr, y, nullptr,
                                                  MTOT, DMOD, DMOD, 0, 0);
        ln_kernel<<<MTOT, 256, 0, stream>>>(y, x, g1 + l * DMOD, b1 + l * DMOD, x, xb);
        gemm_bf16_kernel<<<gFF, 256, 0, stream>>>(xb, w1b + oDF, bf1 + l * DFF, nullptr, hbuf,
                                                  MTOT, DFF, DMOD, 1, 0);
        gemm_bf16_kernel<<<gDD, 256, 0, stream>>>(hbuf, w2b + oDF, bf2 + l * DMOD, y, nullptr,
                                                  MTOT, DMOD, DFF, 0, 0);
        ln_kernel<<<MTOT, 256, 0, stream>>>(y, x, g2 + l * DMOD, b2 + l * DMOD, x, xb);
    }
    head_kernel<<<MTOT, 256, 0, stream>>>(x, hW, hbv, out);
}
